// AttentionModule_34308198761142
// MI455X (gfx1250) — matrix operand order repaired and checked by reading, not compiled, not run on hardware
//
#include <hip/hip_runtime.h>
#include <math.h>

#define EPS 1e-6f
#define B_   4
#define N_   4096
#define TOK  (B_ * N_)

typedef __attribute__((ext_vector_type(16))) _Float16 v16h;
typedef __attribute__((ext_vector_type(8)))  _Float16 v8h;
typedef __attribute__((ext_vector_type(2)))  _Float16 v2h;
typedef __attribute__((ext_vector_type(8)))  float    v8f;

// ---------------- cross-lane helpers ----------------------------------------
__device__ __forceinline__ float xor16_f(float v) { return __shfl_xor(v, 16, 32); }
__device__ __forceinline__ v2h   xor16_h2(v2h v) {
  int r = __shfl_xor(__builtin_bit_cast(int, v), 16, 32);
  return __builtin_bit_cast(v2h, r);
}

// ---------------- per-batch mean of vector norms (batch_norm x-scale) -------
__global__ __launch_bounds__(256) void k_scale(const float* __restrict__ x,
                                               float* __restrict__ scal) {
  int b = blockIdx.x, tid = threadIdx.x;
  const float* xb = x + (size_t)b * N_ * 3;
  float acc = 0.f;
  for (int n = tid; n < N_; n += 256) {
    float a0 = xb[n*3], a1 = xb[n*3+1], a2 = xb[n*3+2];
    acc += sqrtf(a0*a0 + a1*a1 + a2*a2 + EPS);
  }
  __shared__ float red[256];
  red[tid] = acc; __syncthreads();
  for (int s = 128; s > 0; s >>= 1) { if (tid < s) red[tid] += red[tid + s]; __syncthreads(); }
  if (tid == 0) scal[b] = red[0] * (1.f / N_);
}

// ---------------- batch_norm #1 + q/k/v projections -------------------------
//   qh : (B,N,32) row-major f16, softmax scale 1/sqrt(32) folded in
//   kh : (B,N,32) row-major f16
//   vt : (B,32,N) d-major f16 (so V^T A-operands load contiguously)
__global__ __launch_bounds__(256) void k_bn_qkv(
    const float* __restrict__ x, const float* __restrict__ f, const float* __restrict__ scal,
    const float* Wvq, const float* Wiq, const float* biq,
    const float* Wvk, const float* Wik, const float* bik,
    const float* Wvv, const float* Wiv, const float* biv,
    float* __restrict__ qe, float* __restrict__ ke, float* __restrict__ ve,
    _Float16* __restrict__ qh, _Float16* __restrict__ kh, _Float16* __restrict__ vt) {
  __shared__ float sWi[3][1024];
  __shared__ float sb[3][32];
  __shared__ float sWv[3][3];
  int tid = threadIdx.x;
  for (int i = tid; i < 1024; i += 256) { sWi[0][i] = Wiq[i]; sWi[1][i] = Wik[i]; sWi[2][i] = Wiv[i]; }
  if (tid < 32) { sb[0][tid] = biq[tid]; sb[1][tid] = bik[tid]; sb[2][tid] = biv[tid]; }
  if (tid < 3)  { sWv[0][tid] = Wvq[tid]; sWv[1][tid] = Wvk[tid]; sWv[2][tid] = Wvv[tid]; }
  __syncthreads();

  int t = blockIdx.x * 256 + tid;
  int b = t / N_;
  int n = t - b * N_;

  float fr[32]; float mu = 0.f;
#pragma unroll
  for (int d = 0; d < 32; ++d) { fr[d] = f[(size_t)t*32 + d]; mu += fr[d]; }
  mu *= (1.f/32.f);
  float var = 0.f;
#pragma unroll
  for (int d = 0; d < 32; ++d) { float c = fr[d] - mu; var += c*c; }
  var *= (1.f/32.f);
  float rinv = rsqrtf(var + EPS);
#pragma unroll
  for (int d = 0; d < 32; ++d) fr[d] = (fr[d] - mu) * rinv;

  float inv_s = 1.f / (scal[b] + EPS);
  float xh[3];
#pragma unroll
  for (int i = 0; i < 3; ++i) xh[i] = x[(size_t)t*3 + i] * inv_s;

#pragma unroll
  for (int o = 0; o < 3; ++o)
#pragma unroll
    for (int i = 0; i < 3; ++i) {
      qe[(size_t)t*9 + o*3 + i] = xh[i] * sWv[0][o];
      ke[(size_t)t*9 + o*3 + i] = xh[i] * sWv[1][o];
      ve[(size_t)t*9 + o*3 + i] = xh[i] * sWv[2][o];
    }

  const float qscale = 0.17677669529663687f;   // 1/sqrt(32)
#pragma unroll 4
  for (int o = 0; o < 32; ++o) {
    float aq = sb[0][o], ak = sb[1][o], av = sb[2][o];
#pragma unroll
    for (int d = 0; d < 32; ++d) {
      aq = fmaf(fr[d], sWi[0][d*32 + o], aq);
      ak = fmaf(fr[d], sWi[1][d*32 + o], ak);
      av = fmaf(fr[d], sWi[2][d*32 + o], av);
    }
    qh[(size_t)t*32 + o] = (_Float16)(aq * qscale);
    kh[(size_t)t*32 + o] = (_Float16)ak;
    vt[((size_t)b*32 + o) * N_ + n] = (_Float16)av;
  }
}

// ---------------- flash attention (transposed-S formulation) ----------------
// 1 wave = one 16-query tile; 4 waves/block. We compute S^T = K_tile x Q^T so
// each lane holds 16 scores of ONE query; its other 16 keys live in lane^16.
//   - per-block row max  : local v_max3 chain + ONE shfl_xor(16)
//   - softmax denominator: per-lane partials, ONE shuffle at the end
//   - P: exp -> f16 pack -> partner exchange fused per pair (minimal liveness)
//   - O^T = V^T x P^T    : A-operand streams from d-major VT contiguously
__global__ __launch_bounds__(128, 1) void k_attn(const _Float16* __restrict__ Q,
                                                 const _Float16* __restrict__ K,
                                                 const _Float16* __restrict__ VT,
                                                 float* __restrict__ O) {
  const int wave = threadIdx.x >> 5;
  const int lane = threadIdx.x & 31;
  const bool hi  = lane >= 16;
  const int l16  = lane & 15;

  const int tile = blockIdx.x * 4 + wave;    // 0..1023  (B_*N_/16 tiles)
  const int b    = tile >> 8;                // 256 tiles per batch
  const int mrow = (tile & 255) << 4;

  const _Float16* qb  = Q  + (size_t)b * N_ * 32;
  const _Float16* kbp = K  + (size_t)b * N_ * 32;
  const _Float16* vbp = VT + (size_t)b * 32 * N_;

  // Q^T as B operand (32 d x 16 queries): lane = query col, d's 8h..8h+7 and 16+8h..16+8h+7
  v16h qB;
  {
    const _Float16* qr = qb + (size_t)(mrow + l16) * 32 + (hi ? 8 : 0);
    v8h q0 = *(const v8h*)(qr), q1 = *(const v8h*)(qr + 16);
    qB = __builtin_shufflevector(q0, q1, 0,1,2,3,4,5,6,7,8,9,10,11,12,13,14,15);
  }

  // V^T row bases for the two d-halves (A operands, d row = l16 / 16+l16)
  const _Float16* vrow0 = vbp + (size_t)(l16)      * N_;
  const _Float16* vrow1 = vbp + (size_t)(16 + l16) * N_;

  v8f ot0 = {}, ot1 = {};                    // O^T tiles: d x queries
  float m = -1e30f, ps = 0.f;                // per-lane (query l16, 16-key half)

  for (int j = 0; j < N_; j += 32) {
    // A operands: two 16-key tiles of K (16 keys x 32 d), A-layout K-split
    v16h ka0, ka1;
    {
      const _Float16* kr0 = kbp + (size_t)(j + l16)      * 32 + (hi ? 8 : 0);
      const _Float16* kr1 = kbp + (size_t)(j + 16 + l16) * 32 + (hi ? 8 : 0);
      v8h a0 = *(const v8h*)(kr0), a1 = *(const v8h*)(kr0 + 16);
      v8h b0 = *(const v8h*)(kr1), b1 = *(const v8h*)(kr1 + 16);
      ka0 = __builtin_shufflevector(a0, a1, 0,1,2,3,4,5,6,7,8,9,10,11,12,13,14,15);
      ka1 = __builtin_shufflevector(b0, b1, 0,1,2,3,4,5,6,7,8,9,10,11,12,13,14,15);
    }
    v8f zc = {};
    // S^T tiles: st0 = keys j..j+15, st1 = keys j+16..j+31 (x 16 queries)
    v8f st0 = __builtin_amdgcn_wmma_f32_16x16x32_f16(false, ka0, false, qB, (short)0, zc, false, false);
    v8f st1 = __builtin_amdgcn_wmma_f32_16x16x32_f16(false, ka1, false, qB, (short)0, zc, false, false);

    // ---- online softmax: this lane owns 16 of the 32 scores of query l16
    float lm = fmaxf(st0[0], st1[0]);
#pragma unroll
    for (int r = 1; r < 8; ++r) lm = fmaxf(lm, fmaxf(st0[r], st1[r]));
    float tmax = fmaxf(lm, xor16_f(lm));         // combine with partner half
    float mnew = fmaxf(m, tmax);
    float alpha = __expf(m - mnew);
    m = mnew;

    // ---- exp -> f16 pack -> partner exchange, fused per pair (low liveness)
    // lo lane owns tile0 keys 0..7, receives partner's tile0 keys 8..15.
    // hi lane owns tile1 keys 24..31, receives partner's tile1 keys 16..23.
    float lsum = 0.f;
    v8h ownv, rxv;
#pragma unroll
    for (int k = 0; k < 4; ++k) {
      float a0 = __expf(st0[2*k]     - mnew);
      float a1 = __expf(st0[2*k + 1] - mnew);
      float b0 = __expf(st1[2*k]     - mnew);
      float b1 = __expf(st1[2*k + 1] - mnew);
      lsum += (a0 + a1) + (b0 + b1);
      v2h pa2; pa2[0] = (_Float16)a0; pa2[1] = (_Float16)a1;
      v2h pb2; pb2[0] = (_Float16)b0; pb2[1] = (_Float16)b1;
      v2h ow = hi ? pb2 : pa2;
      v2h rc = xor16_h2(hi ? pa2 : pb2);
      ownv[2*k] = ow[0]; ownv[2*k + 1] = ow[1];
      rxv[2*k]  = rc[0]; rxv[2*k + 1]  = rc[1];
    }
    ps = ps * alpha + lsum;
#pragma unroll
    for (int r = 0; r < 8; ++r) { ot0[r] = ot0[r] * alpha; ot1[r] = ot1[r] * alpha; }

    // pB B-layout: lo lane = keys 0..15 (own|rx), hi lane = keys 16..31 (rx|own)
    v16h plo = __builtin_shufflevector(ownv, rxv, 0,1,2,3,4,5,6,7,8,9,10,11,12,13,14,15);
    v16h phi = __builtin_shufflevector(rxv, ownv, 0,1,2,3,4,5,6,7,8,9,10,11,12,13,14,15);
    v16h pB  = hi ? phi : plo;

    // ---- O^T += V^T x P^T : A operands stream contiguously from VT
    v16h va0, va1;
    {
      const _Float16* v0 = vrow0 + j + (hi ? 16 : 0);
      const _Float16* v1 = vrow1 + j + (hi ? 16 : 0);
      v8h a0 = *(const v8h*)(v0), a1 = *(const v8h*)(v0 + 8);
      v8h b0 = *(const v8h*)(v1), b1 = *(const v8h*)(v1 + 8);
      va0 = __builtin_shufflevector(a0, a1, 0,1,2,3,4,5,6,7,8,9,10,11,12,13,14,15);
      va1 = __builtin_shufflevector(b0, b1, 0,1,2,3,4,5,6,7,8,9,10,11,12,13,14,15);
    }
    ot0 = __builtin_amdgcn_wmma_f32_16x16x32_f16(false, va0, false, pB, (short)0, ot0, false, false);
    ot1 = __builtin_amdgcn_wmma_f32_16x16x32_f16(false, va1, false, pB, (short)0, ot1, false, false);

    if (j + 32 < N_) __builtin_prefetch(kbp + (size_t)(j + 32 + l16) * 32, 0, 1);
  }

  // denominator: this lane's partial + partner's partial (one shuffle)
  float inv = 1.f / (ps + xor16_f(ps));
  // O^T layout: lane = query l16, VGPR r = d (r + 8*hi), tile1 = d+16
  float* orow = O + ((size_t)b * N_ + mrow + l16) * 32;
#pragma unroll
  for (int r = 0; r < 8; ++r) {
    orow[r + (hi ? 8 : 0)]      = ot0[r] * inv;
    orow[16 + r + (hi ? 8 : 0)] = ot1[r] * inv;
  }
}

// ---------------- circular convolution (FFT long conv), LDS-resident -------
__global__ __launch_bounds__(256) void k_conv(const float* __restrict__ qe,
                                              const float* __restrict__ ke,
                                              float* __restrict__ uc) {
  __shared__ float qs[N_];
  __shared__ float ks[N_];
  int pair = blockIdx.x;                // 0..35 : (batch, channel)
  int b = pair / 9, c = pair % 9;
  int tid = threadIdx.x;
  for (int i = tid; i < N_; i += 256) {
    qs[i] = qe[((size_t)b * N_ + i) * 9 + c];
    ks[i] = ke[((size_t)b * N_ + i) * 9 + c];
  }
  __syncthreads();
  int n = blockIdx.y * 256 + tid;
  float acc = 0.f;
#pragma unroll 8
  for (int m = 0; m < N_; ++m) acc = fmaf(qs[m], ks[(n - m) & (N_ - 1)], acc);
  uc[((size_t)b * N_ + n) * 9 + c] = acc;
}

// ---------------- gate MLP + cross + channel softmax + p1 proj + residual ---
__global__ __launch_bounds__(256) void k_gate(
    const float* __restrict__ x, const float* __restrict__ f,
    const float* __restrict__ uc, const float* __restrict__ ve,
    const float* __restrict__ ui,
    const float* Wg1, const float* bg1, const float* Wg2, const float* bg2,
    const float* Wvp1, const float* Wip1, const float* bip1,
    float* __restrict__ x1, float* __restrict__ f1) {
  __shared__ float sW1[35 * 64];
  __shared__ float sW2[64 * 35];
  __shared__ float sWi[1024];
  __shared__ float sb1[64], sb2[35], sbi[32], sWv[3];
  int tid = threadIdx.x;
  for (int i = tid; i < 35 * 64; i += 256) { sW1[i] = Wg1[i]; sW2[i] = Wg2[i]; }
  for (int i = tid; i < 1024; i += 256) sWi[i] = Wip1[i];
  if (tid < 64) sb1[tid] = bg1[tid];
  if (tid < 35) sb2[tid] = bg2[tid];
  if (tid < 32) sbi[tid] = bip1[tid];
  if (tid < 3)  sWv[tid] = Wvp1[tid];
  __syncthreads();

  int t = blockIdx.x * 256 + tid;
  float uv[9], vv[9];
#pragma unroll
  for (int i = 0; i < 9; ++i) { uv[i] = uc[(size_t)t*9 + i]; vv[i] = ve[(size_t)t*9 + i]; }
  float uir[32];
#pragma unroll
  for (int d = 0; d < 32; ++d) uir[d] = ui[(size_t)t*32 + d];

  float feats[35];
#pragma unroll
  for (int m = 0; m < 3; ++m)
    feats[m] = sqrtf(uv[m*3]*uv[m*3] + uv[m*3+1]*uv[m*3+1] + uv[m*3+2]*uv[m*3+2] + EPS);
#pragma unroll
  for (int d = 0; d < 32; ++d) feats[3 + d] = uir[d];

  float h[64];
#pragma unroll 2
  for (int o = 0; o < 64; ++o) {
    float a = sb1[o];
#pragma unroll
    for (int d = 0; d < 35; ++d) a = fmaf(feats[d], sW1[d*64 + o], a);
    h[o] = fmaxf(a, 0.f);
  }
  float g[35];
#pragma unroll 2
  for (int o = 0; o < 35; ++o) {
    float a = sb2[o];
#pragma unroll
    for (int d = 0; d < 64; ++d) a = fmaf(h[d], sW2[d*35 + o], a);
    g[o] = a;
  }

  float me[9];
#pragma unroll
  for (int jx = 0; jx < 9; ++jx) me[jx] = g[jx / 3];
  float mx = me[0];
#pragma unroll
  for (int jx = 1; jx < 9; ++jx) mx = fmaxf(mx, me[jx]);
  float se = 0.f, pe[9];
#pragma unroll
  for (int jx = 0; jx < 9; ++jx) { pe[jx] = __expf(me[jx] - mx); se += pe[jx]; }
  float isum = 1.f / se;

  float ue[9];
#pragma unroll
  for (int m = 0; m < 3; ++m) {
    float ux = uv[m*3], uy = uv[m*3+1], uz = uv[m*3+2];
    float vx = vv[m*3], vy = vv[m*3+1], vz = vv[m*3+2];
    ue[m*3]   = pe[m*3]   * isum * (uy*vz - uz*vy);
    ue[m*3+1] = pe[m*3+1] * isum * (uz*vx - ux*vz);
    ue[m*3+2] = pe[m*3+2] * isum * (ux*vy - uy*vx);
  }

#pragma unroll
  for (int i = 0; i < 3; ++i) {
    float a = ue[i]*sWv[0] + ue[3+i]*sWv[1] + ue[6+i]*sWv[2];
    x1[(size_t)t*3 + i] = x[(size_t)t*3 + i] + a;
  }
#pragma unroll 4
  for (int o = 0; o < 32; ++o) {
    float a = sbi[o];
#pragma unroll
    for (int d = 0; d < 32; ++d) a = fmaf(uir[d], sWi[d*32 + o], a);
    f1[(size_t)t*32 + o] = f[(size_t)t*32 + o] + a;
  }
}

// ---------------- batch_norm #2 + norm_act + final projection ---------------
__global__ __launch_bounds__(256) void k_bn_act_proj(
    const float* __restrict__ x1, const float* __restrict__ f1, const float* __restrict__ scal,
    const float* Wvfp, const float* Wifp, const float* bifp,
    float* __restrict__ x3, float* __restrict__ f3) {
  __shared__ float sWi[1024];
  __shared__ float sb[32];
  __shared__ float sWv;
  int tid = threadIdx.x;
  for (int i = tid; i < 1024; i += 256) sWi[i] = Wifp[i];
  if (tid < 32) sb[tid] = bifp[tid];
  if (tid == 0) sWv = Wvfp[0];
  __syncthreads();

  int t = blockIdx.x * 256 + tid;
  int b = t / N_;

  float fr[32]; float mu = 0.f;
#pragma unroll
  for (int d = 0; d < 32; ++d) { fr[d] = f1[(size_t)t*32 + d]; mu += fr[d]; }
  mu *= (1.f/32.f);
  float var = 0.f;
#pragma unroll
  for (int d = 0; d < 32; ++d) { float c = fr[d] - mu; var += c*c; }
  var *= (1.f/32.f);
  float rinv = rsqrtf(var + EPS);
#pragma unroll
  for (int d = 0; d < 32; ++d) fr[d] = fmaxf((fr[d] - mu) * rinv, 0.f);   // bn + relu

  float inv_s = 1.f / (scal[b] + EPS);
  float x0 = x1[(size_t)t*3]   * inv_s;
  float y0 = x1[(size_t)t*3+1] * inv_s;
  float z0 = x1[(size_t)t*3+2] * inv_s;
  float nrm = sqrtf(x0*x0 + y0*y0 + z0*z0 + EPS);
  float sg = 1.f / (1.f + __expf(-nrm));

  x3[(size_t)t*3]   = x0 * sg * sWv;
  x3[(size_t)t*3+1] = y0 * sg * sWv;
  x3[(size_t)t*3+2] = z0 * sg * sWv;

#pragma unroll 4
  for (int o = 0; o < 32; ++o) {
    float a = sb[o];
#pragma unroll
    for (int d = 0; d < 32; ++d) a = fmaf(fr[d], sWi[d*32 + o], a);
    f3[(size_t)t*32 + o] = a;
  }
}

// ---------------- batch_norm #3 + norm_act -> outputs -----------------------
__global__ __launch_bounds__(256) void k_final(const float* __restrict__ x3,
                                               const float* __restrict__ f3,
                                               const float* __restrict__ scal,
                                               float* __restrict__ out) {
  int t = blockIdx.x * 256 + threadIdx.x;
  int b = t / N_;

  float fr[32]; float mu = 0.f;
#pragma unroll
  for (int d = 0; d < 32; ++d) { fr[d] = f3[(size_t)t*32 + d]; mu += fr[d]; }
  mu *= (1.f/32.f);
  float var = 0.f;
#pragma unroll
  for (int d = 0; d < 32; ++d) { float c = fr[d] - mu; var += c*c; }
  var *= (1.f/32.f);
  float rinv = rsqrtf(var + EPS);

  float inv_s = 1.f / (scal[b] + EPS);
  float x0 = x3[(size_t)t*3]   * inv_s;
  float y0 = x3[(size_t)t*3+1] * inv_s;
  float z0 = x3[(size_t)t*3+2] * inv_s;
  float nrm = sqrtf(x0*x0 + y0*y0 + z0*z0 + EPS);
  float sg = 1.f / (1.f + __expf(-nrm));

  out[(size_t)t*3]   = x0 * sg;
  out[(size_t)t*3+1] = y0 * sg;
  out[(size_t)t*3+2] = z0 * sg;
  float* of = out + (size_t)TOK * 3;
#pragma unroll
  for (int d = 0; d < 32; ++d)
    of[(size_t)t*32 + d] = fmaxf((fr[d] - mu) * rinv, 0.f);
}

// ---------------------------------------------------------------------------
extern "C" void kernel_launch(void* const* d_in, const int* in_sizes, int n_in,
                              void* d_out, int out_size, void* d_ws, size_t ws_size,
                              hipStream_t stream) {
  (void)in_sizes; (void)n_in; (void)out_size; (void)ws_size;
  const float* x    = (const float*)d_in[0];
  const float* f    = (const float*)d_in[1];
  const float* Wvq  = (const float*)d_in[2];
  const float* Wiq  = (const float*)d_in[3];
  const float* biq  = (const float*)d_in[4];
  const float* Wvk  = (const float*)d_in[5];
  const float* Wik  = (const float*)d_in[6];
  const float* bik  = (const float*)d_in[7];
  const float* Wvv  = (const float*)d_in[8];
  const float* Wiv  = (const float*)d_in[9];
  const float* biv  = (const float*)d_in[10];
  const float* Wg1  = (const float*)d_in[11];
  const float* bg1  = (const float*)d_in[12];
  const float* Wg2  = (const float*)d_in[13];
  const float* bg2  = (const float*)d_in[14];
  const float* Wvp1 = (const float*)d_in[15];
  const float* Wip1 = (const float*)d_in[16];
  const float* bip1 = (const float*)d_in[17];
  const float* Wvfp = (const float*)d_in[18];
  const float* Wifp = (const float*)d_in[19];
  const float* bifp = (const float*)d_in[20];

  float* w = (float*)d_ws;
  size_t off = 0;
  float* qe = w + off; off += (size_t)TOK * 9;
  float* ke = w + off; off += (size_t)TOK * 9;
  float* ve = w + off; off += (size_t)TOK * 9;
  float* ui = w + off; off += (size_t)TOK * 32;
  float* uc = w + off; off += (size_t)TOK * 9;
  float* x1 = w + off; off += (size_t)TOK * 3;
  float* f1 = w + off; off += (size_t)TOK * 32;
  float* x3 = w + off; off += (size_t)TOK * 3;
  float* f3 = w + off; off += (size_t)TOK * 32;
  float* scal = w + off; off += 16;      // [0..3]=bn1  [4..7]=bn2  [8..11]=bn3
  // f16 region (16B-aligned: all offsets above are multiples of 16 floats)
  _Float16* h = (_Float16*)(w + off);
  size_t hoff = 0;
  _Float16* qh = h + hoff; hoff += (size_t)TOK * 32;
  _Float16* kh = h + hoff; hoff += (size_t)TOK * 32;
  _Float16* vt = h + hoff; hoff += (size_t)TOK * 32;   // (B,32,N) d-major

  k_scale<<<B_, 256, 0, stream>>>(x, scal);
  k_bn_qkv<<<TOK / 256, 256, 0, stream>>>(x, f, scal,
      Wvq, Wiq, biq, Wvk, Wik, bik, Wvv, Wiv, biv,
      qe, ke, ve, qh, kh, vt);
  k_attn<<<(TOK / 16) / 4, 128, 0, stream>>>(qh, kh, vt, ui);
  {
    dim3 gconv(B_ * 9, N_ / 256);
    k_conv<<<gconv, 256, 0, stream>>>(qe, ke, uc);
  }
  k_gate<<<TOK / 256, 256, 0, stream>>>(x, f, uc, ve, ui,
      Wg1, bg1, Wg2, bg2, Wvp1, Wip1, bip1, x1, f1);
  k_scale<<<B_, 256, 0, stream>>>(x1, scal + 4);
  k_bn_act_proj<<<TOK / 256, 256, 0, stream>>>(x1, f1, scal + 4,
      Wvfp, Wifp, bifp, x3, f3);
  k_scale<<<B_, 256, 0, stream>>>(x3, scal + 8);
  k_final<<<TOK / 256, 256, 0, stream>>>(x3, f3, scal + 8, (float*)d_out);
}
